// DGCNNNet_24266565222694
// MI455X (gfx1250) — matrix operand order repaired and checked by reading, not compiled, not run on hardware
//
#include <hip/hip_runtime.h>
#include <cfloat>

// ---------------------------------------------------------------------------
// CDNA5 wave32 WMMA helpers (v_wmma_f32_16x16x32_f16)
// ---------------------------------------------------------------------------
typedef __attribute__((ext_vector_type(16))) _Float16 v16h;
typedef __attribute__((ext_vector_type(8)))  _Float16 v8h;
typedef __attribute__((ext_vector_type(4)))  _Float16 v4h;
typedef __attribute__((ext_vector_type(2)))  _Float16 v2h;
typedef __attribute__((ext_vector_type(8)))  float    v8f;

__device__ __forceinline__ v8f wmma_f32_f16(v16h a, v16h b, v8f c) {
  return __builtin_amdgcn_wmma_f32_16x16x32_f16(
      /*neg_a=*/false, a, /*neg_b=*/false, b,
      /*c_mod=*/(short)0, c, /*reuse_a=*/false, /*reuse_b=*/false);
}

// A fragment: 16x32 f16 tile, row-major in LDS with row stride ldh halves.
// lane<16 -> row=lane, K = k0+[0..7] and k0+[16..23]
// lane>=16 -> row=lane-16, K = k0+[8..15] and k0+[24..31]
__device__ __forceinline__ v16h load_a_frag(const _Float16* tile, int ldh,
                                            int k0, int lane) {
  int row = lane & 15;
  int kb  = k0 + ((lane & 16) ? 8 : 0);
  v8h lo = *(const v8h*)(tile + row * ldh + kb);
  v8h hi = *(const v8h*)(tile + row * ldh + kb + 16);
  v16h a;
#pragma unroll
  for (int i = 0; i < 8; ++i) { a[i] = lo[i]; a[8 + i] = hi[i]; }
  return a;
}

// B fragment: 32x16 f16 tile; stored column-major in LDS (tileT[col*ldh+k]).
__device__ __forceinline__ v16h load_b_frag(const _Float16* tileT, int ldh,
                                            int c0, int k0, int lane) {
  int col = c0 + (lane & 15);
  int kk  = k0 + ((lane & 16) ? 8 : 0);
  v8h lo = *(const v8h*)(tileT + col * ldh + kk);
  v8h hi = *(const v8h*)(tileT + col * ldh + kk + 16);
  v16h b;
#pragma unroll
  for (int i = 0; i < 8; ++i) { b[i] = lo[i]; b[8 + i] = hi[i]; }
  return b;
}

__device__ __forceinline__ v4h cvt4h(float4 a) {
  v4h h = {(_Float16)a.x, (_Float16)a.y, (_Float16)a.z, (_Float16)a.w};
  return h;
}
__device__ __forceinline__ void store_pair(_Float16* dst, float a, float b) {
  v2h p = {(_Float16)a, (_Float16)b};
  *(v2h*)dst = p;   // dst at even half index -> 4B aligned -> ds_store_b32
}

// ---------------------------------------------------------------------------
// Tensor Data Mover: 2D f32 tile (rows x cols, row stride in elems) -> LDS.
// 6-arg builtin form (clang-23 / therock-10.0):
//   (uint32x4 g0, int32x8 g1, int32x4 g2, int32x4 g3, int32x8 g4, i32 cpol)
// D# per CDNA5 ISA.
// ---------------------------------------------------------------------------
#if defined(__has_builtin)
#if __has_builtin(__builtin_amdgcn_tensor_load_to_lds)
#define HAVE_TDM 1
#endif
#endif

#ifdef HAVE_TDM
typedef __attribute__((ext_vector_type(4))) unsigned int u32x4;
typedef __attribute__((ext_vector_type(8))) int          i32x8;
typedef __attribute__((ext_vector_type(4))) int          i32x4;

__device__ __forceinline__ void tdm_load_tile_f32(unsigned lds_off,
                                                  const float* gsrc,
                                                  unsigned rows, unsigned cols,
                                                  unsigned strideElems) {
  unsigned long long ga = (unsigned long long)(size_t)gsrc;
  u32x4 g0;
  g0[0] = 1u;                                    // count=1 (user descriptor)
  g0[1] = lds_off;                               // lds_addr (bytes)
  g0[2] = (unsigned)(ga & 0xFFFFFFFFu);          // global_addr[31:0]
  g0[3] = (unsigned)((ga >> 32) & 0x1FFFFFFu)    // global_addr[56:32]
          | (2u << 30);                          // type=2 ("image")
  i32x8 g1;
  g1[0] = (int)(2u << 16);                       // wg_mask=0, data_size=4B
  g1[1] = (int)(cols << 16);                     // tensor_dim0[15:0]
  g1[2] = (int)((cols >> 16) | (rows << 16));    // dim0 hi | tensor_dim1 lo
  g1[3] = (int)((rows >> 16) | (cols << 16));    // dim1 hi | tile_dim0
  g1[4] = (int)(rows & 0xFFFFu);                 // tile_dim1, tile_dim2=0
  g1[5] = (int)strideElems;                      // tensor_dim0_stride lo
  g1[6] = 0;                                     // stride hi | dim1_stride lo
  g1[7] = 0;
  i32x4 gz4 = {0, 0, 0, 0};
  i32x8 gz8 = {0, 0, 0, 0, 0, 0, 0, 0};
  __builtin_amdgcn_tensor_load_to_lds(g0, g1, gz4, gz4, gz8, 0);
}
#endif

// ---------------------------------------------------------------------------
// 0) concat x,pos -> x0 (16384 x 6)
// ---------------------------------------------------------------------------
__global__ void concat_kernel(const float* __restrict__ x,
                              const float* __restrict__ pos,
                              float* __restrict__ x0, int n) {
  int i = blockIdx.x * 256 + threadIdx.x;
  if (i < n) {
#pragma unroll
    for (int k = 0; k < 3; ++k) {
      x0[i * 6 + k]     = x[i * 3 + k];
      x0[i * 6 + 3 + k] = pos[i * 3 + k];
    }
  }
}

// ---------------------------------------------------------------------------
// 1) KNN: 30 nearest (incl. self) within each batch of 2048 points.
// ---------------------------------------------------------------------------
template <int D>
__global__ void __launch_bounds__(128)
knn_kernel(const float* __restrict__ f, int ld, int* __restrict__ knn) {
  __shared__ __align__(16) float sF[128 * D];
  const int tid = threadIdx.x;
  const int gp  = blockIdx.x * 128 + tid;
  const int rowBase = (gp >> 11) << 11;   // batch base (N = 2048)

  float fi[D];
#pragma unroll
  for (int k = 0; k < D; ++k) fi[k] = f[gp * ld + k];

  float dist[30];
  int   idx[30];
#pragma unroll
  for (int s = 0; s < 30; ++s) { dist[s] = FLT_MAX; idx[s] = rowBase; }

  for (int jb = 0; jb < 2048; jb += 128) {
    if constexpr ((D & 3) == 0) {
#pragma unroll
      for (int k4 = 0; k4 < D; k4 += 4) {
        float4 v = *(const float4*)&f[(rowBase + jb + tid) * ld + k4];
        *(float4*)&sF[tid * D + k4] = v;
      }
    } else {
      for (int k = 0; k < D; ++k)
        sF[tid * D + k] = f[(rowBase + jb + tid) * ld + k];
    }
    __syncthreads();
    for (int j = 0; j < 128; ++j) {
      float d = 0.f;
#pragma unroll
      for (int k = 0; k < D; ++k) {
        float df = fi[k] - sF[j * D + k];
        d += df * df;
      }
      if (d < dist[29]) {
        int pos = 29;
        while (pos > 0 && dist[pos - 1] > d) {
          dist[pos] = dist[pos - 1];
          idx[pos]  = idx[pos - 1];
          --pos;
        }
        dist[pos] = d;
        idx[pos]  = rowBase + jb + j;
      }
    }
    __syncthreads();
  }
  for (int s = 0; s < 30; ++s) knn[gp * 30 + s] = idx[s];
}

// ---------------------------------------------------------------------------
// 2) EdgeConv: e=[xi, xj-xi] (2*DIN, padded to KP) -> 64 relu -> 64,
//    max over 30 neighbors. Block = 16 points, 8 waves.
// ---------------------------------------------------------------------------
template <int DIN, int KP>
__global__ void __launch_bounds__(256)
edgeconv_kernel(const float* __restrict__ fin, int ldin,
                const int* __restrict__ knn,
                const float* __restrict__ wa, const float* __restrict__ ba,
                const float* __restrict__ wb, const float* __restrict__ bb,
                float* __restrict__ fout, int ldout) {
  __shared__ __align__(16) _Float16 sWaT[64 * KP];  // wa^T col-major, k-padded
  __shared__ __align__(16) _Float16 sWbT[64 * 64];  // wb^T col-major
  __shared__ __align__(16) _Float16 sE[16 * KP];    // edge tile (f16)
  __shared__ __align__(16) _Float16 sH[16 * 64];    // hidden after relu (f16)
  __shared__ float sMax[16 * 64];
  __shared__ float sBa[64], sBb[64];

  const int tid  = threadIdx.x;
  const int lane = tid & 31;
  const int wave = tid >> 5;
  const int p0   = blockIdx.x * 16;
  const float4 f4z = make_float4(0.f, 0.f, 0.f, 0.f);

  // wa^T: pair-pack (k,k+1) per column, zero-pad k >= 2*DIN
  for (int q = tid; q < 16 * (KP / 2); q += 256) {
    int c4 = (q & 15) * 4;
    int kk = (q >> 4) * 2;
    float4 r0 = (kk     < 2 * DIN) ? *(const float4*)&wa[(kk    ) * 64 + c4] : f4z;
    float4 r1 = (kk + 1 < 2 * DIN) ? *(const float4*)&wa[(kk + 1) * 64 + c4] : f4z;
    store_pair(&sWaT[(c4 + 0) * KP + kk], r0.x, r1.x);
    store_pair(&sWaT[(c4 + 1) * KP + kk], r0.y, r1.y);
    store_pair(&sWaT[(c4 + 2) * KP + kk], r0.z, r1.z);
    store_pair(&sWaT[(c4 + 3) * KP + kk], r0.w, r1.w);
  }
  // wb^T
  for (int q = tid; q < 16 * 32; q += 256) {
    int c4 = (q & 15) * 4;
    int kk = (q >> 4) * 2;
    float4 r0 = *(const float4*)&wb[(kk    ) * 64 + c4];
    float4 r1 = *(const float4*)&wb[(kk + 1) * 64 + c4];
    store_pair(&sWbT[(c4 + 0) * 64 + kk], r0.x, r1.x);
    store_pair(&sWbT[(c4 + 1) * 64 + kk], r0.y, r1.y);
    store_pair(&sWbT[(c4 + 2) * 64 + kk], r0.z, r1.z);
    store_pair(&sWbT[(c4 + 3) * 64 + kk], r0.w, r1.w);
  }
  if (tid < 64) { sBa[tid] = ba[tid]; sBb[tid] = bb[tid]; }
  for (int e = tid; e < 16 * 64; e += 256) sMax[e] = -FLT_MAX;
  __syncthreads();

  for (int t = 0; t < 32; ++t) {
    const int pl    = t >> 1;
    const int gp    = p0 + pl;
    const int sbase = (t & 1) * 16;

    if constexpr (DIN == 64 && KP == 128) {
      // vector staging: 16 rows x 32 float4-groups = 512 chunks
      for (int q = tid; q < 512; q += 256) {
        int r  = q >> 5;
        int k4 = (q & 31) * 4;
        int s  = sbase + r;
        if (s >= 30) s -= 30;
        v4h h;
        if (k4 < 64) {
          float4 a = *(const float4*)&fin[gp * ldin + k4];
          h = cvt4h(a);
        } else {
          int j = knn[gp * 30 + s];
          float4 aj = *(const float4*)&fin[j  * ldin + (k4 - 64)];
          float4 ai = *(const float4*)&fin[gp * ldin + (k4 - 64)];
          h = cvt4h(make_float4(aj.x - ai.x, aj.y - ai.y,
                                aj.z - ai.z, aj.w - ai.w));
        }
        *(v4h*)&sE[r * KP + k4] = h;
      }
    } else {
      for (int e = tid; e < 16 * KP; e += 256) {
        int r = e / KP, k = e % KP;
        int s = sbase + r;
        if (s >= 30) s -= 30;
        float val = 0.f;
        if (k < DIN) {
          val = fin[gp * ldin + k];
        } else if (k < 2 * DIN) {
          int j  = knn[gp * 30 + s];
          int kk = k - DIN;
          val = fin[j * ldin + kk] - fin[gp * ldin + kk];
        }
        sE[e] = (_Float16)val;
      }
    }
    __syncthreads();

    if (wave < 4) {                    // GEMM1: e @ wa + ba, relu -> sH
      const int nc = wave * 16;
      v8f acc = {};
#pragma unroll
      for (int k0 = 0; k0 < KP; k0 += 32) {
        v16h a = load_a_frag(sE, KP, k0, lane);
        v16h b = load_b_frag(sWaT, KP, nc, k0, lane);
        acc = wmma_f32_f16(a, b, acc);
      }
      const int   c  = nc + (lane & 15);
      const int   rb = (lane & 16) ? 8 : 0;
      const float bv = sBa[c];
#pragma unroll
      for (int v = 0; v < 8; ++v) {
        float h = fmaxf(acc[v] + bv, 0.f);
        sH[(rb + v) * 64 + c] = (_Float16)h;
      }
    }
    __syncthreads();

    if (wave >= 4) {                   // GEMM2: h @ wb + bb, max-fold
      const int nc = (wave - 4) * 16;
      v8f acc = {};
#pragma unroll
      for (int k0 = 0; k0 < 64; k0 += 32) {
        v16h a = load_a_frag(sH, 64, k0, lane);
        v16h b = load_b_frag(sWbT, 64, nc, k0, lane);
        acc = wmma_f32_f16(a, b, acc);
      }
      const int   c  = nc + (lane & 15);
      const float bv = sBb[c];
      float m = -FLT_MAX;
#pragma unroll
      for (int v = 0; v < 8; ++v) m = fmaxf(m, acc[v] + bv);
      m = fmaxf(m, __shfl_xor(m, 16, 32));
      if (lane < 16) sMax[pl * 64 + c] = fmaxf(sMax[pl * 64 + c], m);
    }
    __syncthreads();
  }

  for (int e = tid; e < 16 * 64; e += 256) {
    int r = e >> 6, c = e & 63;
    fout[(p0 + r) * ldout + c] = sMax[e];
  }
}

// ---------------------------------------------------------------------------
// 3) Head GEMM: Y = act(A(MxK) @ W(KxN) + bias). 64x64 block tile, 8 waves.
//    W chunk staged via Tensor Data Mover (raw f32 tile -> LDS), then
//    converted/transposed to f16 col-major from LDS.
// ---------------------------------------------------------------------------
__global__ void __launch_bounds__(256)
gemm_kernel(const float* __restrict__ A, const float* __restrict__ W,
            const float* __restrict__ bias, float* __restrict__ out,
            int M, int Kdim, int N, int relu) {
  __shared__ __align__(16) _Float16 sA[64 * 32];    // A chunk, row-major f16
  __shared__ __align__(16) _Float16 sBT[64 * 32];   // W chunk, col-major f16
#ifdef HAVE_TDM
  __shared__ __align__(16) float sWstage[32 * 64];  // raw f32 W tile (TDM dst)
#endif

  const int tid    = threadIdx.x;
  const int lane   = tid & 31;
  const int wave   = tid >> 5;
  const int mBlock = blockIdx.y * 64;
  const int nBlock = blockIdx.x * 64;

  const int mr0 = ((wave    ) >> 2) * 16, nc0 = ((wave    ) & 3) * 16;
  const int mr1 = ((wave + 8) >> 2) * 16, nc1 = ((wave + 8) & 3) * 16;
  v8f acc0 = {}, acc1 = {};

  for (int k0 = 0; k0 < Kdim; k0 += 32) {
    // ---- stage A chunk: 64x32 f32 -> f16, b128 global loads
    for (int q = tid; q < 512; q += 256) {
      int r  = q >> 3;
      int c4 = (q & 7) * 4;
      float4 a = *(const float4*)&A[(mBlock + r) * Kdim + k0 + c4];
      *(v4h*)&sA[r * 32 + c4] = cvt4h(a);
    }

#ifdef HAVE_TDM
    // ---- stage W chunk via TDM: 32 rows x 64 cols f32 tile
    if (wave == 0) {
      tdm_load_tile_f32((unsigned)(size_t)&sWstage[0],
                        W + (size_t)k0 * N + nBlock, 32u, 64u, (unsigned)N);
      __builtin_amdgcn_s_wait_tensorcnt(0);
    }
    __syncthreads();
    // convert/transpose from LDS stage: sBT[c*32+kk] = sWstage[kk*64+c]
    for (int q = tid; q < 256; q += 256) {
      int c4 = (q & 15) * 4;
      int kk = (q >> 4) * 2;
      float4 r0 = *(const float4*)&sWstage[(kk    ) * 64 + c4];
      float4 r1 = *(const float4*)&sWstage[(kk + 1) * 64 + c4];
      store_pair(&sBT[(c4 + 0) * 32 + kk], r0.x, r1.x);
      store_pair(&sBT[(c4 + 1) * 32 + kk], r0.y, r1.y);
      store_pair(&sBT[(c4 + 2) * 32 + kk], r0.z, r1.z);
      store_pair(&sBT[(c4 + 3) * 32 + kk], r0.w, r1.w);
    }
#else
    // ---- fallback: direct global pair-pack transpose
    for (int q = tid; q < 256; q += 256) {
      int c4 = (q & 15) * 4;
      int kk = (q >> 4) * 2;
      float4 r0 = *(const float4*)&W[(k0 + kk    ) * N + nBlock + c4];
      float4 r1 = *(const float4*)&W[(k0 + kk + 1) * N + nBlock + c4];
      store_pair(&sBT[(c4 + 0) * 32 + kk], r0.x, r1.x);
      store_pair(&sBT[(c4 + 1) * 32 + kk], r0.y, r1.y);
      store_pair(&sBT[(c4 + 2) * 32 + kk], r0.z, r1.z);
      store_pair(&sBT[(c4 + 3) * 32 + kk], r0.w, r1.w);
    }
#endif
    if (k0 + 32 < Kdim) {
      __builtin_prefetch(&A[(mBlock + (tid & 63)) * Kdim + k0 + 32], 0, 0);
    }
    __syncthreads();

    {
      v16h a = load_a_frag(sA + mr0 * 32, 32, 0, lane);
      v16h b = load_b_frag(sBT, 32, nc0, 0, lane);
      acc0 = wmma_f32_f16(a, b, acc0);
    }
    {
      v16h a = load_a_frag(sA + mr1 * 32, 32, 0, lane);
      v16h b = load_b_frag(sBT, 32, nc1, 0, lane);
      acc1 = wmma_f32_f16(a, b, acc1);
    }
    __syncthreads();
  }

  const int rb = (lane & 16) ? 8 : 0;
  {
    int   c  = nc0 + (lane & 15);
    float bv = bias[nBlock + c];
#pragma unroll
    for (int v = 0; v < 8; ++v) {
      float y = acc0[v] + bv;
      if (relu) y = fmaxf(y, 0.f);
      out[(mBlock + mr0 + rb + v) * N + nBlock + c] = y;
    }
  }
  {
    int   c  = nc1 + (lane & 15);
    float bv = bias[nBlock + c];
#pragma unroll
    for (int v = 0; v < 8; ++v) {
      float y = acc1[v] + bv;
      if (relu) y = fmaxf(y, 0.f);
      out[(mBlock + mr1 + rb + v) * N + nBlock + c] = y;
    }
  }
}

// ---------------------------------------------------------------------------
// launch
// ---------------------------------------------------------------------------
extern "C" void kernel_launch(void* const* d_in, const int* in_sizes, int n_in,
                              void* d_out, int out_size, void* d_ws,
                              size_t ws_size, hipStream_t stream) {
  const float* x   = (const float*)d_in[0];
  const float* pos = (const float*)d_in[1];
  const float* w1a = (const float*)d_in[2];
  const float* b1a = (const float*)d_in[3];
  const float* w1b = (const float*)d_in[4];
  const float* b1b = (const float*)d_in[5];
  const float* w2a = (const float*)d_in[6];
  const float* b2a = (const float*)d_in[7];
  const float* w2b = (const float*)d_in[8];
  const float* b2b = (const float*)d_in[9];
  const float* w3a = (const float*)d_in[10];
  const float* b3a = (const float*)d_in[11];
  const float* w3b = (const float*)d_in[12];
  const float* b3b = (const float*)d_in[13];
  const float* hw1 = (const float*)d_in[14];
  const float* hb1 = (const float*)d_in[15];
  const float* hw2 = (const float*)d_in[16];
  const float* hb2 = (const float*)d_in[17];
  const float* hw3 = (const float*)d_in[18];
  const float* hb3 = (const float*)d_in[19];

  const int NP = 16384;                 // B*N
  float* ws   = (float*)d_ws;
  float* x0   = ws;                     // 16384*6
  float* feat = x0 + NP * 6;            // 16384*192 (x1|x2|x3)
  float* h1   = feat + NP * 192;        // 16384*1024
  float* h2   = h1 + NP * 1024;         // 16384*512
  int*   kidx = (int*)(h2 + NP * 512);  // 16384*30

  concat_kernel<<<NP / 256, 256, 0, stream>>>(x, pos, x0, NP);

  knn_kernel<6><<<NP / 128, 128, 0, stream>>>(x0, 6, kidx);
  edgeconv_kernel<6, 32><<<NP / 16, 256, 0, stream>>>(
      x0, 6, kidx, w1a, b1a, w1b, b1b, feat + 0, 192);

  knn_kernel<64><<<NP / 128, 128, 0, stream>>>(feat + 0, 192, kidx);
  edgeconv_kernel<64, 128><<<NP / 16, 256, 0, stream>>>(
      feat + 0, 192, kidx, w2a, b2a, w2b, b2b, feat + 64, 192);

  knn_kernel<64><<<NP / 128, 128, 0, stream>>>(feat + 64, 192, kidx);
  edgeconv_kernel<64, 128><<<NP / 16, 256, 0, stream>>>(
      feat + 64, 192, kidx, w3a, b3a, w3b, b3b, feat + 128, 192);

  gemm_kernel<<<dim3(1024 / 64, NP / 64), 256, 0, stream>>>(
      feat, hw1, hb1, h1, NP, 192, 1024, 1);
  gemm_kernel<<<dim3(512 / 64, NP / 64), 256, 0, stream>>>(
      h1, hw2, hb2, h2, NP, 1024, 512, 1);
  gemm_kernel<<<dim3(128 / 64, NP / 64), 256, 0, stream>>>(
      h2, hw3, hb3, (float*)d_out, NP, 512, 128, 0);
}
